// CriticGraphPolicy_71665824301174
// MI455X (gfx1250) — compile-verified, matrix operand order checked by reading, not run on hardware
//
#include <hip/hip_runtime.h>

typedef _Float16 f16;
typedef __attribute__((ext_vector_type(16))) _Float16 v16h;
typedef __attribute__((ext_vector_type(8)))  float    v8f;
typedef __attribute__((ext_vector_type(4)))  unsigned int v4u;
typedef __attribute__((ext_vector_type(8)))  int v8i;
typedef __attribute__((ext_vector_type(4)))  int v4i;

#define TM   64      // rows per Q block
#define K1   64      // padded effective input dim (39 real)
#define N1   400     // layer-1 width (25 tiles of 16)
#define K2P  416     // padded K for layer 2 (13 chunks of 32)
#define N2P  304     // padded layer-2 width (19 tiles of 16)
#define N2   300

// ---------------------------------------------------------------------------
// WMMA helpers (CDNA5 16x16x32 f16, f32 accumulate)
// ---------------------------------------------------------------------------
static __device__ __forceinline__ v8f wmma32(v16h a, v16h b, v8f c) {
  // (neg_a, A, neg_b, B, c_mod, C, reuse_a, reuse_b)
  return __builtin_amdgcn_wmma_f32_16x16x32_f16(false, a, false, b, (short)0, c,
                                                false, false);
}

// Load a 16x32 fragment (A: row-major MxK, or B from N-major transposed
// weights so K pairs are contiguous). Per ISA 7.12.2 (16-bit A 16x32):
//   lane L: r = L&15, kh = 8*(L>>4); VGPR j   = elems [kh+2j, kh+2j+1]
//                                    VGPR 4+j = elems [16+kh+2j, 16+kh+2j+1]
static __device__ __forceinline__ v16h load_frag(const f16* tile, int stride,
                                                 int lane) {
  const int r  = lane & 15;
  const int kh = (lane >> 4) << 3;
  const f16* p = tile + r * stride + kh;
  v16h v;
#pragma unroll
  for (int j = 0; j < 4; ++j) {
    v[2 * j]     = p[2 * j];
    v[2 * j + 1] = p[2 * j + 1];
    v[2 * j + 8] = p[2 * j + 16];
    v[2 * j + 9] = p[2 * j + 17];
  }
  return v;
}

// ---------------------------------------------------------------------------
// Tensor Data Mover: DMA a 2-D tile (rows of tile_d0 8-byte elements,
// tile_d1 rows, row stride stride0 elements) from global memory into LDS.
// D# layout per CDNA5 ISA (groups 0/1; 2-D => groups
// 2/3 zero). Tracked with TENSORcnt. Issue from ONE wave only.
// ---------------------------------------------------------------------------
static __device__ __forceinline__ void tdm_load_2d(const void* gptr,
                                                   unsigned lds_addr,
                                                   unsigned tensor_d0,
                                                   unsigned tensor_d1,
                                                   unsigned tile_d0,
                                                   unsigned tile_d1,
                                                   unsigned stride0) {
  const unsigned long long ga = (unsigned long long)(uintptr_t)gptr;
  v4u g0;
  g0[0] = 1u;                                    // count=1, user desc
  g0[1] = lds_addr;                              // LDS byte address
  g0[2] = (unsigned)ga;                          // global addr [31:0]
  g0[3] = (unsigned)((ga >> 32) & 0x01FFFFFFu)   // global addr [56:32]
          | (2u << 30);                          // type = 2 ("image")
  v8i g1;
  g1[0] = (int)(3u << 16);                       // data_size = 3 (8 bytes)
  g1[1] = (int)((tensor_d0 & 0xFFFFu) << 16);    // tensor_dim0[15:0]
  g1[2] = (int)(((tensor_d1 & 0xFFFFu) << 16) |  // tensor_dim1[15:0]
                ((tensor_d0 >> 16) & 0xFFFFu));  // tensor_dim0[31:16]
  g1[3] = (int)(((tile_d0 & 0xFFFFu) << 16) |    // tile_dim0
                ((tensor_d1 >> 16) & 0xFFFFu));  // tensor_dim1[31:16]
  g1[4] = (int)(tile_d1 & 0xFFFFu);              // tile_dim1 (tile_dim2=0)
  g1[5] = (int)stride0;                          // tensor_dim0_stride[31:0]
  g1[6] = 0;                                     // stride0[47:32], stride1 lo
  g1[7] = 0;
  const v4i gz = {0, 0, 0, 0};
#if __clang_major__ >= 23
  const v8i gz8 = {0, 0, 0, 0, 0, 0, 0, 0};
  __builtin_amdgcn_tensor_load_to_lds(g0, g1, gz, gz, gz8, 0);
#else
  __builtin_amdgcn_tensor_load_to_lds(g0, g1, gz, gz, 0);
#endif
}

// ---------------------------------------------------------------------------
// Kernel 1: repack q-net weights into f16, transposed, zero-row-folded
// ---------------------------------------------------------------------------
__global__ __launch_bounds__(256) void prep_kernel(
    const float* __restrict__ q1w1, const float* __restrict__ q2w1,
    const float* __restrict__ q1w2, const float* __restrict__ q2w2,
    const float* __restrict__ q1b1, const float* __restrict__ q2b1,
    const float* __restrict__ q1b2, const float* __restrict__ q2b2,
    f16* __restrict__ W1t, f16* __restrict__ W2t,
    float* __restrict__ b1f, float* __restrict__ b2f) {
  const int R1 = 2 * N1 * K1;    // 51200
  const int R2 = 2 * N2P * K2P;  // 252928
  const int R3 = 2 * N1;
  const int R4 = 2 * N2P;
  int idx = blockIdx.x * 256 + threadIdx.x;
  if (idx < R1) {
    int net = idx / (N1 * K1);
    int rem = idx % (N1 * K1);
    int n = rem / K1, k = rem % K1;
    const float* w = net ? q2w1 : q1w1;
    // xum rows 6..44 of q_w1 are the only ones multiplying nonzeros
    W1t[idx] = (k < 39) ? (f16)w[(6 + k) * N1 + n] : (f16)0.f;
  } else if (idx < R1 + R2) {
    int i = idx - R1;
    int net = i / (N2P * K2P);
    int rem = i % (N2P * K2P);
    int n = rem / K2P, k = rem % K2P;
    const float* w = net ? q2w2 : q1w2;
    W2t[i] = (n < N2 && k < N1) ? (f16)w[k * N2 + n] : (f16)0.f;
  } else if (idx < R1 + R2 + R3) {
    int i = idx - R1 - R2;
    int net = i / N1, n = i % N1;
    b1f[i] = (net ? q2b1 : q1b1)[n];
  } else if (idx < R1 + R2 + R3 + R4) {
    int i = idx - R1 - R2 - R3;
    int net = i / N2P, n = i % N2P;
    b2f[i] = (n < N2) ? (net ? q2b2 : q1b2)[n] : 0.f;
  }
}

// ---------------------------------------------------------------------------
// Kernel 2: upstream net (f32, ~1% of FLOPs). One thread per batch row.
// Emits packed f16 input row: [pos,pos,msg_up(32),action, 0-pad] (64 wide)
// ---------------------------------------------------------------------------
__global__ __launch_bounds__(256) void upstream_kernel(
    const float* __restrict__ state, const float* __restrict__ action,
    const float* __restrict__ w1, const float* __restrict__ b1,
    const float* __restrict__ w2, const float* __restrict__ b2,
    const float* __restrict__ w3, const float* __restrict__ b3,
    f16* __restrict__ Xin, int B) {
  __shared__ float sW1[33 * 64];
  __shared__ float sW2[64 * 64];  // only rows 0..63 of up_fc2_w matter
  __shared__ float sW3[64 * 32];
  __shared__ float sb1[64], sb2[64], sb3[32];
  const int tid = threadIdx.x;
  for (int i = tid; i < 33 * 64; i += 256) sW1[i] = w1[i];
  for (int i = tid; i < 64 * 64; i += 256) sW2[i] = w2[i];
  for (int i = tid; i < 64 * 32; i += 256) sW3[i] = w3[i];
  if (tid < 64) { sb1[tid] = b1[tid]; sb2[tid] = b2[tid]; }
  if (tid < 32) sb3[tid] = b3[tid];
  __syncthreads();

  const int b = blockIdx.x * 256 + tid;
  if (b >= B) return;

  float xin[33];
  const float* srow = state + (size_t)b * 32;
#pragma unroll 8
  for (int k = 0; k < 32; ++k) xin[k] = srow[k];
  const float act = action[b];
  xin[32] = act;

  float t[64];
  for (int n = 0; n < 64; ++n) t[n] = sb1[n];
  for (int k = 0; k < 33; ++k) {
    const float v = xin[k];
#pragma unroll 8
    for (int n = 0; n < 64; ++n) t[n] += v * sW1[k * 64 + n];
  }
  float ss = 0.f;
  for (int n = 0; n < 64; ++n) ss += t[n] * t[n];
  float inv = 1.f / fmaxf(sqrtf(ss), 1e-12f);

  float h[64];
  for (int n = 0; n < 64; ++n) h[n] = tanhf(t[n] * inv);

  float t2[64];
  for (int n = 0; n < 64; ++n) t2[n] = sb2[n];
  for (int k = 0; k < 64; ++k) {
    const float v = h[k];
#pragma unroll 8
    for (int n = 0; n < 64; ++n) t2[n] += v * sW2[k * 64 + n];
  }
  for (int n = 0; n < 64; ++n) h[n] = tanhf(t2[n]);

  float t3[32];
  for (int n = 0; n < 32; ++n) t3[n] = sb3[n];
  for (int k = 0; k < 64; ++k) {
    const float v = h[k];
#pragma unroll 8
    for (int n = 0; n < 32; ++n) t3[n] += v * sW3[k * 32 + n];
  }
  ss = 0.f;
  for (int n = 0; n < 32; ++n) ss += t3[n] * t3[n];
  inv = 1.f / fmaxf(sqrtf(ss), 1e-12f);

  union { f16 hv[64]; uint4 q[8]; } u;
#pragma unroll
  for (int n = 0; n < 64; ++n) u.hv[n] = (f16)0.f;
  u.hv[0] = (f16)srow[0]; u.hv[1] = (f16)srow[1]; u.hv[2] = (f16)srow[2];
  u.hv[3] = u.hv[0];      u.hv[4] = u.hv[1];      u.hv[5] = u.hv[2];
#pragma unroll
  for (int n = 0; n < 32; ++n) u.hv[6 + n] = (f16)(t3[n] * inv);
  u.hv[38] = (f16)act;

  uint4* dst = (uint4*)(Xin + (size_t)b * K1);
#pragma unroll
  for (int j = 0; j < 8; ++j) dst[j] = u.q[j];
}

// ---------------------------------------------------------------------------
// Kernel 3: fused Q MLP (64x400 -> ReLU -> 400x300 -> ReLU -> dot w3) via
// WMMA, with TDM (tensor_load_to_lds) staging of the input tile and
// double-buffered layer-2 weight chunks.
// grid = (B/64, 2 nets), 256 threads = 8 waves.
// wave w: M-tile = w&3 (16 rows), N-half = w>>2.
// ---------------------------------------------------------------------------
__global__ __launch_bounds__(256) void qnet_kernel(
    const f16* __restrict__ Xin, const f16* __restrict__ W1t,
    const f16* __restrict__ W2t, const float* __restrict__ b1f,
    const float* __restrict__ b2f,
    const float* __restrict__ w3_1, const float* __restrict__ b3_1,
    const float* __restrict__ w3_2, const float* __restrict__ b3_2,
    float* __restrict__ out, int B) {
  __shared__ alignas(16) f16 Xs[TM * K1];          //  8 KB (TDM dest)
  __shared__ alignas(16) f16 H1s[TM * K2P];        // 52 KB
  __shared__ alignas(16) f16 Wc[2 * N2P * 32];     // 38 KB (TDM double buf)
  __shared__ alignas(16) f16 H2s[TM * N2P];        // 38 KB
  __shared__ float psum[256];

  const int tid  = threadIdx.x;
  const int lane = tid & 31;
  const int wv   = tid >> 5;
  const int mt   = wv & 3;   // M tile (16 rows)
  const int nh   = wv >> 2;  // N half
  const int net  = blockIdx.y;
  const int row0 = blockIdx.x * TM;

  const f16* W2n = W2t + (size_t)net * N2P * K2P;

  // Issue TDM DMAs: input tile (1-D, 8KB contiguous) and layer-2 chunk 0.
  // The chunk-0 DMA overlaps all of layer 1.
  if (wv == 0) {
    tdm_load_2d(Xin + (size_t)row0 * K1, (unsigned)(uintptr_t)&Xs[0],
                /*tensor_d0=*/TM * K1 / 4, /*tensor_d1=*/1,
                /*tile_d0=*/TM * K1 / 4, /*tile_d1=*/0,
                /*stride0=*/TM * K1 / 4);
    tdm_load_2d(W2n, (unsigned)(uintptr_t)&Wc[0],
                /*tensor_d0=*/K2P / 4, /*tensor_d1=*/N2P,
                /*tile_d0=*/32 / 4, /*tile_d1=*/N2P,
                /*stride0=*/K2P / 4);
    __builtin_amdgcn_s_wait_tensorcnt(1);  // Xs complete (in-order)
  }
  // Zero the K-pad columns (400..415) of H1 while the DMAs fly
  for (int i = tid; i < TM * 16 / 2; i += 256) {
    int r = i >> 3, c = i & 7;
    ((unsigned*)(H1s + r * K2P + N1))[c] = 0u;
  }
  __syncthreads();

  // ---------------- Layer 1: (64x64) @ (64x400) + b1, ReLU -> H1s ----------
  const f16* W1n = W1t + (size_t)net * N1 * K1;
  const v16h a0 = load_frag(Xs + mt * 16 * K1, K1, lane);
  const v16h a1 = load_frag(Xs + mt * 16 * K1 + 32, K1, lane);
  const int nt0 = nh * 12;  // halves: 0..12 and 12..24 (tile 12 overlaps)
  for (int i = 0; i < 13; ++i) {
    const int nt = nt0 + i;
    v8f acc = {0.f, 0.f, 0.f, 0.f, 0.f, 0.f, 0.f, 0.f};
    const v16h b0 = load_frag(W1n + (size_t)nt * 16 * K1, K1, lane);
    const v16h b1v = load_frag(W1n + (size_t)nt * 16 * K1 + 32, K1, lane);
    acc = wmma32(a0, b0, acc);
    acc = wmma32(a1, b1v, acc);
    if (!(nh && i == 0)) {  // skip duplicate store of overlap tile
      const int col = nt * 16 + (lane & 15);
      const float bias = b1f[net * N1 + col];
      const int rb = mt * 16 + ((lane >> 4) << 3);
#pragma unroll
      for (int r = 0; r < 8; ++r) {
        H1s[(rb + r) * K2P + col] = (f16)fmaxf(acc[r] + bias, 0.f);
      }
    }
  }
  __syncthreads();

  // ---------------- Layer 2: (64x400) @ (400x300) + b2, ReLU -> H2s --------
  v8f acc2[10];
#pragma unroll
  for (int t = 0; t < 10; ++t)
    acc2[t] = (v8f){0.f, 0.f, 0.f, 0.f, 0.f, 0.f, 0.f, 0.f};
  const int ntb = nh * 9;  // halves: tiles 0..9 and 9..18 (tile 9 overlaps)
  for (int kc = 0; kc < 13; ++kc) {
    if (wv == 0) {
      if (kc + 1 < 13) {
        // Prefetch next chunk into the other buffer via TDM, then wait for
        // the current chunk (in-order => <=1 outstanding leaves kc done).
        tdm_load_2d(W2n + (kc + 1) * 32,
                    (unsigned)(uintptr_t)&Wc[((kc + 1) & 1) * N2P * 32],
                    K2P / 4, N2P, 32 / 4, N2P, K2P / 4);
        __builtin_amdgcn_s_wait_tensorcnt(1);
      } else {
        __builtin_amdgcn_s_wait_tensorcnt(0);
      }
    }
    __syncthreads();
    const f16* WcBuf = Wc + (kc & 1) * N2P * 32;
    const v16h a = load_frag(H1s + mt * 16 * K2P + kc * 32, K2P, lane);
#pragma unroll
    for (int t = 0; t < 10; ++t) {
      const v16h b = load_frag(WcBuf + (ntb + t) * 16 * 32, 32, lane);
      acc2[t] = wmma32(a, b, acc2[t]);
    }
    __syncthreads();
  }
#pragma unroll
  for (int t = 0; t < 10; ++t) {
    if (nh && t == 0) continue;  // skip duplicate overlap tile
    const int col = (ntb + t) * 16 + (lane & 15);
    const float bias = b2f[net * N2P + col];
    const int rb = mt * 16 + ((lane >> 4) << 3);
#pragma unroll
    for (int r = 0; r < 8; ++r) {
      H2s[(rb + r) * N2P + col] = (f16)fmaxf(acc2[t][r] + bias, 0.f);
    }
  }
  __syncthreads();

  // ---------------- Layer 3: dot(H2 row, w3) + b3 --------------------------
  const float* w3 = net ? w3_2 : w3_1;
  const float* b3 = net ? b3_2 : b3_1;
  {
    const int row = tid >> 2, part = tid & 3;
    const f16* hrow = H2s + row * N2P;
    float s = 0.f;
    for (int n = part * 75; n < part * 75 + 75; ++n)
      s += (float)hrow[n] * w3[n];
    psum[tid] = s;
  }
  __syncthreads();
  if (tid < TM) {
    const float v = psum[4 * tid] + psum[4 * tid + 1] + psum[4 * tid + 2] +
                    psum[4 * tid + 3] + b3[0];
    out[(size_t)net * B + row0 + tid] = v;
  }
}

// ---------------------------------------------------------------------------
extern "C" void kernel_launch(void* const* d_in, const int* in_sizes, int n_in,
                              void* d_out, int out_size, void* d_ws,
                              size_t ws_size, hipStream_t stream) {
  const float* state  = (const float*)d_in[0];
  const float* action = (const float*)d_in[1];
  const float* up_w1 = (const float*)d_in[2];
  const float* up_b1 = (const float*)d_in[3];
  const float* up_w2 = (const float*)d_in[4];
  const float* up_b2 = (const float*)d_in[5];
  const float* up_w3 = (const float*)d_in[6];
  const float* up_b3 = (const float*)d_in[7];
  // d_in[8], d_in[9] = att_w/att_b -> multiply zeros, unused
  const float* q1w1 = (const float*)d_in[10];
  const float* q1b1 = (const float*)d_in[11];
  const float* q1w2 = (const float*)d_in[12];
  const float* q1b2 = (const float*)d_in[13];
  const float* q1w3 = (const float*)d_in[14];
  const float* q1b3 = (const float*)d_in[15];
  const float* q2w1 = (const float*)d_in[16];
  const float* q2b1 = (const float*)d_in[17];
  const float* q2w2 = (const float*)d_in[18];
  const float* q2b2 = (const float*)d_in[19];
  const float* q2w3 = (const float*)d_in[20];
  const float* q2b3 = (const float*)d_in[21];
  // d_in[22..27] = msg net -> dead code in reference

  const int B = in_sizes[0] / 32;

  char* ws = (char*)d_ws;
  size_t off = 0;
  f16* Xin = (f16*)(ws + off); off += (size_t)B * K1 * sizeof(f16);
  f16* W1t = (f16*)(ws + off); off += (size_t)2 * N1 * K1 * sizeof(f16);
  f16* W2t = (f16*)(ws + off); off += (size_t)2 * N2P * K2P * sizeof(f16);
  float* b1f = (float*)(ws + off); off += (size_t)2 * N1 * sizeof(float);
  float* b2f = (float*)(ws + off); off += (size_t)2 * N2P * sizeof(float);
  (void)ws_size;

  float* out = (float*)d_out;

  const int prep_total = 2 * N1 * K1 + 2 * N2P * K2P + 2 * N1 + 2 * N2P;
  prep_kernel<<<(prep_total + 255) / 256, 256, 0, stream>>>(
      q1w1, q2w1, q1w2, q2w2, q1b1, q2b1, q1b2, q2b2, W1t, W2t, b1f, b2f);

  upstream_kernel<<<(B + 255) / 256, 256, 0, stream>>>(
      state, action, up_w1, up_b1, up_w2, up_b2, up_w3, up_b3, Xin, B);

  dim3 grid(B / TM, 2);
  qnet_kernel<<<grid, 256, 0, stream>>>(Xin, W1t, W2t, b1f, b2f, q1w3, q1b3,
                                        q2w3, q2b3, out, B);
}
